// ReformerWithCustomEmbeddings_82042465288282
// MI455X (gfx1250) — matrix operand order repaired and checked by reading, not compiled, not run on hardware
//
#include <hip/hip_runtime.h>

// ---------------------------------------------------------------- constants
#define B_   8
#define S_   4096
#define D_   512
#define L_   4
#define H_   8
#define DH_  64
#define FF_  2048
#define CH_  64
#define NC_  (S_ / CH_)      // 64 chunks
#define ROWS (B_ * S_)       // 32768

typedef __attribute__((ext_vector_type(16))) __bf16 v16bf;
typedef __attribute__((ext_vector_type(8)))  float  v8f;

union BF16Frag { uint4 q[2]; unsigned short u[16]; v16bf v; };

static __device__ __forceinline__ unsigned short f2bfu(float f) {
  unsigned u = __builtin_bit_cast(unsigned, f);
  unsigned r = (u + 0x7FFFu + ((u >> 16) & 1u)) >> 16;   // round-to-nearest-even
  return (unsigned short)r;
}

// A-fragment K index for 16x16x32 bf16 WMMA (contiguous k = 16*half + s, matching B)
static __device__ __forceinline__ int kbaseA(int j, int half) {
  return 2 * j + 16 * half;
}
// B-fragment K index (16-bit B layout, 32x16)
static __device__ __forceinline__ int kbaseB(int j, int half) {
  return 2 * j + 16 * half;
}

// ---------------------------------------------------------------- weight swizzle:
// fp32 [K,N] -> bf16 in B-fragment-major layout.
// Fragment (kb, nb) covers k in [32kb,32kb+32), n in [16nb,16nb+16); 512 elems.
// Within a fragment, lane = (n&15) + 16*((k>>4)&1) owns 16 contiguous elems,
// position (k&15)  (== 2j+which of the WMMA B operand).
__global__ void swizzle_w_k(const float* __restrict__ src,
                            unsigned short* __restrict__ dst, int K, int N) {
  size_t layer = (size_t)blockIdx.y * K * N;
  int i = blockIdx.x * 256 + threadIdx.x;
  if (i >= K * N) return;
  int k = i / N, n = i - k * N;
  int kb = k >> 5, kr = k & 31;
  int nb = n >> 4, nr = n & 15;
  int lane = nr + 16 * ((kr >> 4) & 1);
  size_t off = ((size_t)nb * (K >> 5) + kb) * 512 + lane * 16 + (kr & 15);
  dst[layer + off] = f2bfu(src[layer + i]);
}

// ---------------------------------------------------------------- embedding
__global__ void embed_k(const int* __restrict__ ids, const float* __restrict__ emb,
                        float* __restrict__ x1, float* __restrict__ x2) {
  size_t i = (size_t)blockIdx.x * 256 + threadIdx.x;   // over ROWS * 128 float4
  int d4 = (int)(i & 127);
  size_t tok = i >> 7;
  int id = ids[tok];
  float4 v = *(const float4*)&emb[(size_t)id * D_ + d4 * 4];
  ((float4*)x1)[i] = v;
  ((float4*)x2)[i] = v;
}

// ---------------------------------------------------------------- LayerNorm (D=512) -> bf16
__global__ __launch_bounds__(128) void layernorm_bf16_k(
    const float* __restrict__ x, const float* __restrict__ sc,
    const float* __restrict__ bi, unsigned short* __restrict__ out) {
  int row = blockIdx.x;
  int tid = threadIdx.x, lane = tid & 31, wave = tid >> 5;
  const float* xr = x + (size_t)row * D_;
  float4 v = ((const float4*)xr)[tid];
  float s = v.x + v.y + v.z + v.w;
  float ss = v.x * v.x + v.y * v.y + v.z * v.z + v.w * v.w;
#pragma unroll
  for (int off = 1; off < 32; off <<= 1) {
    s  += __shfl_xor(s,  off, 32);
    ss += __shfl_xor(ss, off, 32);
  }
  __shared__ float rs[4], rss[4];
  if (lane == 0) { rs[wave] = s; rss[wave] = ss; }
  __syncthreads();
  s  = rs[0]  + rs[1]  + rs[2]  + rs[3];
  ss = rss[0] + rss[1] + rss[2] + rss[3];
  float mean = s * (1.0f / D_);
  float var  = ss * (1.0f / D_) - mean * mean;
  float inv  = rsqrtf(var + 1e-12f);
  int c = tid * 4;
  float e[4] = {v.x, v.y, v.z, v.w};
#pragma unroll
  for (int j = 0; j < 4; ++j)
    out[(size_t)row * D_ + c + j] = f2bfu((e[j] - mean) * inv * sc[c + j] + bi[c + j]);
}

// ---------------------------------------------------------------- final LayerNorm over concat(x1,x2) -> fp32
__global__ __launch_bounds__(256) void final_ln_k(
    const float* __restrict__ x1, const float* __restrict__ x2,
    const float* __restrict__ sc, const float* __restrict__ bi,
    float* __restrict__ out) {
  int row = blockIdx.x;
  int tid = threadIdx.x, lane = tid & 31, wave = tid >> 5;
  int c = tid * 4;
  const float* src = (c < D_) ? (x1 + (size_t)row * D_ + c)
                              : (x2 + (size_t)row * D_ + (c - D_));
  float4 v = *(const float4*)src;
  float s = v.x + v.y + v.z + v.w;
  float ss = v.x * v.x + v.y * v.y + v.z * v.z + v.w * v.w;
#pragma unroll
  for (int off = 1; off < 32; off <<= 1) {
    s  += __shfl_xor(s,  off, 32);
    ss += __shfl_xor(ss, off, 32);
  }
  __shared__ float rs[8], rss[8];
  if (lane == 0) { rs[wave] = s; rss[wave] = ss; }
  __syncthreads();
  s = 0.f; ss = 0.f;
#pragma unroll
  for (int w = 0; w < 8; ++w) { s += rs[w]; ss += rss[w]; }
  float mean = s * (1.0f / (2 * D_));
  float var  = ss * (1.0f / (2 * D_)) - mean * mean;
  float inv  = rsqrtf(var + 1e-12f);
  float e[4] = {v.x, v.y, v.z, v.w};
#pragma unroll
  for (int j = 0; j < 4; ++j)
    out[(size_t)row * (2 * D_) + c + j] = (e[j] - mean) * inv * sc[c + j] + bi[c + j];
}

// ---------------------------------------------------------------- WMMA GEMM
// A: [M,K] bf16 row-major -> LDS tile (grouped b128 loads, staggered waits)
// Bw: pre-swizzled fragment-major bf16 (direct from global/L2, 2x b128 per frag)
// K-tile = 64: 16 WMMAs per barrier pair.
__global__ __launch_bounds__(256) void gemm_bf16_k(
    const unsigned short* __restrict__ A,
    const unsigned short* __restrict__ Bw,
    float* __restrict__ Cf, unsigned short* __restrict__ Cb,
    const float* __restrict__ bias, const float* __restrict__ res,
    int M, int N, int K, int relu) {
  __shared__ __align__(16) unsigned short As[128][64];   // 16 KB
  const int tid = threadIdx.x;
  const int lane = tid & 31;
  const int wave = tid >> 5;          // 8 waves
  const int wm = wave >> 2;           // 0..1 (M)
  const int wn = wave & 3;            // 0..3 (N)
  const int bm = blockIdx.x * 128;
  const int bn = blockIdx.y * 128;
  const int r = lane & 15, half = lane >> 4;
  const int kblocks = K >> 5;
  const size_t nbBase = (size_t)((bn >> 4) + wn * 2);
  const int ldRow0 = tid >> 3, ldCol = (tid & 7) * 8;    // + t*32 rows

  v8f acc[4][2];
  v8f zero = {0.f, 0.f, 0.f, 0.f, 0.f, 0.f, 0.f, 0.f};
#pragma unroll
  for (int mt = 0; mt < 4; ++mt)
#pragma unroll
    for (int nt = 0; nt < 2; ++nt) acc[mt][nt] = zero;

  for (int k0 = 0; k0 < K; k0 += 64) {
    // A tile: issue all 4 global loads, then commit (waits stagger)
    {
      uint4 t0 = *(const uint4*)&A[(size_t)(bm + ldRow0 +  0) * K + k0 + ldCol];
      uint4 t1 = *(const uint4*)&A[(size_t)(bm + ldRow0 + 32) * K + k0 + ldCol];
      uint4 t2 = *(const uint4*)&A[(size_t)(bm + ldRow0 + 64) * K + k0 + ldCol];
      uint4 t3 = *(const uint4*)&A[(size_t)(bm + ldRow0 + 96) * K + k0 + ldCol];
      *(uint4*)&As[ldRow0 +  0][ldCol] = t0;
      *(uint4*)&As[ldRow0 + 32][ldCol] = t1;
      *(uint4*)&As[ldRow0 + 64][ldCol] = t2;
      *(uint4*)&As[ldRow0 + 96][ldCol] = t3;
    }
    if (k0 + 64 < K) {  // warm L2/L0 for the next tile
      __builtin_prefetch(&A[(size_t)(bm + ldRow0) * K + k0 + 64 + ldCol], 0, 1);
      __builtin_prefetch(&A[(size_t)(bm + ldRow0 + 64) * K + k0 + 64 + ldCol], 0, 1);
    }
    __syncthreads();

    // B fragments straight from global (L2-resident weights)
    BF16Frag bf[2][2];
#pragma unroll
    for (int kb2 = 0; kb2 < 2; ++kb2)
#pragma unroll
      for (int nt = 0; nt < 2; ++nt) {
        const unsigned short* p =
            Bw + ((nbBase + nt) * kblocks + (k0 >> 5) + kb2) * 512 + lane * 16;
        bf[kb2][nt].q[0] = ((const uint4*)p)[0];
        bf[kb2][nt].q[1] = ((const uint4*)p)[1];
        if (k0 + 64 < K) __builtin_prefetch(p + 1024, 0, 1);
      }
#pragma unroll
    for (int kb2 = 0; kb2 < 2; ++kb2) {
      BF16Frag af[4];
#pragma unroll
      for (int mt = 0; mt < 4; ++mt) {
        int row = wm * 64 + mt * 16 + r;
#pragma unroll
        for (int j = 0; j < 8; ++j) {
          int kbi = kb2 * 32 + kbaseA(j, half);
          af[mt].u[2 * j]     = As[row][kbi];
          af[mt].u[2 * j + 1] = As[row][kbi + 1];
        }
      }
#pragma unroll
      for (int mt = 0; mt < 4; ++mt)
#pragma unroll
        for (int nt = 0; nt < 2; ++nt)
          acc[mt][nt] = __builtin_amdgcn_wmma_f32_16x16x32_bf16(
              false, af[mt].v, false, bf[kb2][nt].v, (short)0, acc[mt][nt],
              false, false);
    }
    __syncthreads();
  }

#pragma unroll
  for (int mt = 0; mt < 4; ++mt)
#pragma unroll
    for (int nt = 0; nt < 2; ++nt) {
      int colg = bn + wn * 32 + nt * 16 + r;
      float bv = bias ? bias[colg] : 0.0f;
#pragma unroll
      for (int rr = 0; rr < 8; ++rr) {
        int rowg = bm + wm * 64 + mt * 16 + rr + 8 * half;
        float v = acc[mt][nt][rr] + bv;
        if (relu) v = fmaxf(v, 0.0f);
        if (res) v += res[(size_t)rowg * N + colg];
        if (Cf) Cf[(size_t)rowg * N + colg] = v;
        if (Cb) Cb[(size_t)rowg * N + colg] = f2bfu(v);
      }
    }
}

// ---------------------------------------------------------------- LSH rotations (hash-based Gaussians)
__global__ void gen_rot_k(float* __restrict__ rot, int n) {
  int i = blockIdx.x * 256 + threadIdx.x;
  if (i >= n) return;
  unsigned a = (unsigned)i * 0x9E3779B1u + 0x85EBCA77u;
  a ^= a >> 15; a *= 0x2C1B3C6Du; a ^= a >> 12; a *= 0x297A2D39u; a ^= a >> 15;
  unsigned b = ((unsigned)i ^ 0xDEADBEEFu) * 0x85EBCA77u + 0xC2B2AE3Du;
  b ^= b >> 13; b *= 0x9E3779B1u; b ^= b >> 16;
  float u1 = ((a >> 8) + 1u) * (1.0f / 16777216.0f);
  float u2 = (b >> 8) * (1.0f / 16777216.0f);
  rot[i] = sqrtf(-2.0f * __logf(u1)) * __cosf(6.28318530718f * u2);
}

// ---------------------------------------------------------------- bucket assignment -> composite sort keys
__global__ __launch_bounds__(256) void lsh_bucket_k(
    const float* __restrict__ qk, const float* __restrict__ rot,
    unsigned* __restrict__ keys) {
  int idx = blockIdx.x * 256 + threadIdx.x;   // ((b*H+h)<<12)+s
  int s = idx & (S_ - 1);
  int h = (idx >> 12) & (H_ - 1);
  int b = idx >> 15;
  const float* q = qk + ((size_t)(b * S_ + s) * (H_ * DH_)) + h * DH_;
  float qv[64];
#pragma unroll
  for (int t = 0; t < 16; ++t) ((float4*)qv)[t] = ((const float4*)q)[t];
  const float* R = rot + (size_t)h * DH_ * 32;
  float best = -3.4e38f; int bi = 0;
  for (int rI = 0; rI < 32; ++rI) {
    float acc = 0.0f;
#pragma unroll
    for (int d = 0; d < 64; ++d) acc = fmaf(qv[d], R[d * 32 + rI], acc);
    if (acc > best)  { best = acc;  bi = rI; }
    if (-acc > best) { best = -acc; bi = rI + 32; }
  }
  keys[idx] = (unsigned)(bi * S_ + s);
}

// ---------------------------------------------------------------- bitonic sort of 4096 keys per (b,h) in LDS
__global__ __launch_bounds__(1024) void sort_k(unsigned* __restrict__ keys) {
  __shared__ unsigned sk[4096];
  unsigned base = (unsigned)blockIdx.x << 12;
  int tid = threadIdx.x;
  for (int i = tid; i < 4096; i += 1024) sk[i] = keys[base + i];
  __syncthreads();
  for (unsigned k = 2; k <= 4096; k <<= 1) {
    for (unsigned j = k >> 1; j > 0; j >>= 1) {
      for (unsigned t = tid; t < 2048; t += 1024) {
        unsigned i = 2 * t - (t & (j - 1));
        unsigned p = i | j;
        bool up = ((i & k) == 0);
        unsigned a = sk[i], c = sk[p];
        if (up ? (a > c) : (a < c)) { sk[i] = c; sk[p] = a; }
      }
      __syncthreads();
    }
  }
  for (int i = tid; i < 4096; i += 1024) keys[base + i] = sk[i];
}

// ---------------------------------------------------------------- chunked LSH attention with WMMA
// K-matrix is stored in LDS directly in B-fragment layout (ksw):
// fragment id = (d>>5)*8 + (key>>4); within it lane=(key&15)+16*((d>>4)&1),
// position d&15 -> writer (owning a key row) emits 4 contiguous 32B chunks,
// and the score matmul loads each fragment as two ds_load_b128.
__global__ __launch_bounds__(128) void lsh_attn_k(
    const float* __restrict__ qk,          // [B,S,H*DH] fp32
    const unsigned short* __restrict__ vb, // [B,S,H*DH] bf16
    const unsigned* __restrict__ skeys,    // [B,H,S] sorted bucket*S+pos
    unsigned short* __restrict__ out) {    // [B,S,H*DH] bf16, scattered to orig pos
  __shared__ __align__(16) unsigned short qs[64][64];     // 8 KB
  __shared__ __align__(16) unsigned short ksw[16 * 512];  // 16 KB (2 kb x 8 nb frags)
  __shared__ __align__(16) unsigned short vs[128][64];    // 16 KB
  __shared__ __align__(16) unsigned short ps[64][128];    // 16 KB
  __shared__ int kpos[128];

  const int chunk = blockIdx.x, h = blockIdx.y, b = blockIdx.z;
  const int tid = threadIdx.x, lane = tid & 31, wave = tid >> 5;
  const int r = lane & 15, half = lane >> 4;

  { // gather sorted rows: 0..63 = previous chunk, 64..127 = this chunk
    int i = tid;
    int prev = (chunk + NC_ - 1) & (NC_ - 1);
    int gi = ((i < 64) ? prev : chunk) * CH_ + (i & 63);
    unsigned key = skeys[(((b * H_) + h) << 12) + gi];
    int p = (int)(key & (S_ - 1));
    kpos[i] = p;
    const float* qrow = qk + ((size_t)(b * S_ + p) * (H_ * DH_)) + h * DH_;
    const unsigned short* vrow = vb + ((size_t)(b * S_ + p) * (H_ * DH_)) + h * DH_;
    float qv[64];
#pragma unroll
    for (int t = 0; t < 16; ++t) ((float4*)qv)[t] = ((const float4*)qrow)[t];
#pragma unroll
    for (int t = 0; t < 8; ++t) ((uint4*)&vs[i][0])[t] = ((const uint4*)vrow)[t];
    float ssq = 0.0f;
#pragma unroll
    for (int d = 0; d < 64; ++d) ssq = fmaf(qv[d], qv[d], ssq);
    float inv = 1.0f / fmaxf(sqrtf(ssq), 1e-6f);
    // normalized K -> swizzled fragment layout (4 contiguous 32B chunks)
    int nb = i >> 4, nl = i & 15;
#pragma unroll
    for (int c = 0; c < 4; ++c) {         // c = kb*2 + khalf
      int kb = c >> 1, kh = c & 1;
      BF16Frag tmp;
#pragma unroll
      for (int t = 0; t < 16; ++t)
        tmp.u[t] = f2bfu(qv[kb * 32 + kh * 16 + t] * inv);
      unsigned short* dstp = &ksw[(kb * 8 + nb) * 512 + (nl + 16 * kh) * 16];
      ((uint4*)dstp)[0] = tmp.q[0];
      ((uint4*)dstp)[1] = tmp.q[1];
    }
    if (i >= 64) {
#pragma unroll
      for (int d = 0; d < 64; ++d) qs[i - 64][d] = f2bfu(qv[d] * 0.125f); // 1/sqrt(dh)
    }
  }
  __syncthreads();

  // ---- scores S[64,128] = q @ k^T (each wave: one 16-row M-tile x 8 N-tiles)
  v8f sacc[8];
  v8f zero = {0.f, 0.f, 0.f, 0.f, 0.f, 0.f, 0.f, 0.f};
#pragma unroll
  for (int nt = 0; nt < 8; ++nt) sacc[nt] = zero;
#pragma unroll
  for (int kk = 0; kk < 64; kk += 32) {
    BF16Frag af;
    int row = wave * 16 + r;
#pragma unroll
    for (int j = 0; j < 8; ++j) {
      int kbi = kbaseA(j, half) + kk;
      af.u[2 * j]     = qs[row][kbi];
      af.u[2 * j + 1] = qs[row][kbi + 1];
    }
#pragma unroll
    for (int nt = 0; nt < 8; ++nt) {
      BF16Frag bf;
      const unsigned short* p = &ksw[((kk >> 5) * 8 + nt) * 512 + lane * 16];
      bf.q[0] = ((const uint4*)p)[0];
      bf.q[1] = ((const uint4*)p)[1];
      sacc[nt] = __builtin_amdgcn_wmma_f32_16x16x32_bf16(
          false, af.v, false, bf.v, (short)0, sacc[nt], false, false);
    }
  }

  // ---- self-mask + softmax (rows live in one VGPR slot across a 16-lane group)
  int qrowpos[8];
#pragma unroll
  for (int rr = 0; rr < 8; ++rr)
    qrowpos[rr] = kpos[64 + wave * 16 + rr + 8 * half];
  float rowmax[8];
#pragma unroll
  for (int rr = 0; rr < 8; ++rr) rowmax[rr] = -3.4e38f;
#pragma unroll
  for (int nt = 0; nt < 8; ++nt) {
    int kc = kpos[nt * 16 + r];
#pragma unroll
    for (int rr = 0; rr < 8; ++rr) {
      float s = sacc[nt][rr];
      if (kc == qrowpos[rr]) s = -1e5f;     // SELF_MASK
      sacc[nt][rr] = s;
      rowmax[rr] = fmaxf(rowmax[rr], s);
    }
  }
#pragma unroll
  for (int rr = 0; rr < 8; ++rr)
#pragma unroll
    for (int off = 1; off < 16; off <<= 1)
      rowmax[rr] = fmaxf(rowmax[rr], __shfl_xor(rowmax[rr], off, 32));
  float rowsum[8];
#pragma unroll
  for (int rr = 0; rr < 8; ++rr) rowsum[rr] = 0.0f;
#pragma unroll
  for (int nt = 0; nt < 8; ++nt)
#pragma unroll
    for (int rr = 0; rr < 8; ++rr) {
      float e = __expf(sacc[nt][rr] - rowmax[rr]);
      sacc[nt][rr] = e;
      rowsum[rr] += e;
    }
#pragma unroll
  for (int rr = 0; rr < 8; ++rr)
#pragma unroll
    for (int off = 1; off < 16; off <<= 1)
      rowsum[rr] += __shfl_xor(rowsum[rr], off, 32);
#pragma unroll
  for (int nt = 0; nt < 8; ++nt) {
    int col = nt * 16 + r;
#pragma unroll
    for (int rr = 0; rr < 8; ++rr)
      ps[wave * 16 + rr + 8 * half][col] = f2bfu(sacc[nt][rr] / rowsum[rr]);
  }
  __syncthreads();

  // ---- O[64,64] = P[64,128] @ V[128,64]
  v8f oacc[4];
#pragma unroll
  for (int nt = 0; nt < 4; ++nt) oacc[nt] = zero;
#pragma unroll
  for (int kk = 0; kk < 128; kk += 32) {
    BF16Frag af;
    int row = wave * 16 + r;
#pragma unroll
    for (int j = 0; j < 8; ++j) {
      int kbi = kbaseA(j, half) + kk;
      af.u[2 * j]     = ps[row][kbi];
      af.u[2 * j + 1] = ps[row][kbi + 1];
    }
#pragma unroll
    for (int nt = 0; nt < 4; ++nt) {
      BF16Frag bf;
      int col = nt * 16 + r;
#pragma unroll
      for (int j = 0; j < 8; ++j) {
        int kbi = kbaseB(j, half) + kk;
        bf.u[2 * j]     = vs[kbi][col];
        bf.u[2 * j + 1] = vs[kbi + 1][col];
      }
      oacc[nt] = __builtin_amdgcn_wmma_f32_16x16x32_bf16(
          false, af.v, false, bf.v, (short)0, oacc[nt], false, false);
    }
  }

  // ---- scatter to original positions (spos == sorted key & (S-1))
#pragma unroll
  for (int nt = 0; nt < 4; ++nt) {
    int col = nt * 16 + r;
#pragma unroll
    for (int rr = 0; rr < 8; ++rr) {
      int row = wave * 16 + rr + 8 * half;
      int p = kpos[64 + row];
      out[((size_t)(b * S_ + p) * (H_ * DH_)) + h * DH_ + col] = f2bfu(oacc[nt][rr]);
    }
  }
}

// ---------------------------------------------------------------- host launcher
extern "C" void kernel_launch(void* const* d_in, const int* in_sizes, int n_in,
                              void* d_out, int out_size, void* d_ws, size_t ws_size,
                              hipStream_t stream) {
  (void)in_sizes; (void)n_in; (void)out_size; (void)ws_size;
  const int*   ids    = (const int*)d_in[0];
  const float* emb    = (const float*)d_in[1];
  const float* Wqk    = (const float*)d_in[2];
  const float* Wv     = (const float*)d_in[3];
  const float* Wo     = (const float*)d_in[4];
  const float* ln_a_s = (const float*)d_in[5];
  const float* ln_a_b = (const float*)d_in[6];
  const float* W1     = (const float*)d_in[7];
  const float* b1     = (const float*)d_in[8];
  const float* W2     = (const float*)d_in[9];
  const float* b2     = (const float*)d_in[10];
  const float* ln_f_s = (const float*)d_in[11];
  const float* ln_f_b = (const float*)d_in[12];
  const float* ln_o_s = (const float*)d_in[13];
  const float* ln_o_b = (const float*)d_in[14];
  float* out = (float*)d_out;

  char* ws = (char*)d_ws;
  size_t off = 0;
  auto alloc = [&](size_t bytes) -> char* {
    char* p = ws + off;
    off += (bytes + 255) & ~(size_t)255;
    return p;
  };
  const size_t nWqk = (size_t)L_ * D_ * (H_ * DH_);
  const size_t nW1  = (size_t)L_ * D_ * FF_;
  unsigned short* wqk_bf = (unsigned short*)alloc(nWqk * 2);
  unsigned short* wv_bf  = (unsigned short*)alloc(nWqk * 2);
  unsigned short* wo_bf  = (unsigned short*)alloc(nWqk * 2);
  unsigned short* w1_bf  = (unsigned short*)alloc(nW1 * 2);
  unsigned short* w2_bf  = (unsigned short*)alloc(nW1 * 2);
  float* x1          = (float*)alloc((size_t)ROWS * D_ * 4);
  float* x2          = (float*)alloc((size_t)ROWS * D_ * 4);
  unsigned short* lnbuf   = (unsigned short*)alloc((size_t)ROWS * D_ * 2);
  float* qkbuf       = (float*)alloc((size_t)ROWS * D_ * 4);
  unsigned short* vbuf    = (unsigned short*)alloc((size_t)ROWS * D_ * 2);
  unsigned short* attnout = (unsigned short*)alloc((size_t)ROWS * D_ * 2);
  unsigned short* mid     = (unsigned short*)alloc((size_t)ROWS * FF_ * 2);
  unsigned* keys     = (unsigned*)alloc((size_t)B_ * H_ * S_ * 4);
  float* rot         = (float*)alloc((size_t)L_ * H_ * DH_ * 32 * 4);

  // weights -> bf16, swizzled to WMMA B-fragment layout
  swizzle_w_k<<<dim3((D_ * 512 + 255) / 256, L_), 256, 0, stream>>>(Wqk, wqk_bf, D_, 512);
  swizzle_w_k<<<dim3((D_ * 512 + 255) / 256, L_), 256, 0, stream>>>(Wv,  wv_bf,  D_, 512);
  swizzle_w_k<<<dim3((D_ * 512 + 255) / 256, L_), 256, 0, stream>>>(Wo,  wo_bf,  512, D_);
  swizzle_w_k<<<dim3((D_ * FF_ + 255) / 256, L_), 256, 0, stream>>>(W1, w1_bf, D_, FF_);
  swizzle_w_k<<<dim3((FF_ * D_ + 255) / 256, L_), 256, 0, stream>>>(W2, w2_bf, FF_, D_);

  // LSH rotations for all layers
  const int nrot = L_ * H_ * DH_ * 32;
  gen_rot_k<<<(nrot + 255) / 256, 256, 0, stream>>>(rot, nrot);

  // embedding -> both reversible streams
  embed_k<<<(ROWS * (D_ / 4)) / 256, 256, 0, stream>>>(ids, emb, x1, x2);

  const dim3 gProj(ROWS / 128, D_ / 128);    // N=512
  const dim3 gFF1(ROWS / 128, FF_ / 128);    // N=2048
  for (int l = 0; l < L_; ++l) {
    // attention LN on x2
    layernorm_bf16_k<<<ROWS, 128, 0, stream>>>(x2, ln_a_s + l * D_, ln_a_b + l * D_, lnbuf);
    // qk / v projections
    gemm_bf16_k<<<gProj, 256, 0, stream>>>(lnbuf, wqk_bf + (size_t)l * D_ * 512,
                                           qkbuf, nullptr, nullptr, nullptr,
                                           ROWS, 512, 512, 0);
    gemm_bf16_k<<<gProj, 256, 0, stream>>>(lnbuf, wv_bf + (size_t)l * D_ * 512,
                                           nullptr, vbuf, nullptr, nullptr,
                                           ROWS, 512, 512, 0);
    // bucket + stable sort (composite key = bucket*S + pos)
    lsh_bucket_k<<<(B_ * H_ * S_) / 256, 256, 0, stream>>>(qkbuf, rot + (size_t)l * H_ * DH_ * 32, keys);
    sort_k<<<B_ * H_, 1024, 0, stream>>>(keys);
    // chunked attention, scattered back to original positions
    lsh_attn_k<<<dim3(NC_, H_, B_), 128, 0, stream>>>(qkbuf, vbuf, keys, attnout);
    // y1 = x1 + attn @ Wo   (in place on x1)
    gemm_bf16_k<<<gProj, 256, 0, stream>>>(attnout, wo_bf + (size_t)l * 512 * D_,
                                           x1, nullptr, nullptr, x1,
                                           ROWS, 512, 512, 0);
    // FFN LN on y1
    layernorm_bf16_k<<<ROWS, 128, 0, stream>>>(x1, ln_f_s + l * D_, ln_f_b + l * D_, lnbuf);
    // mid = relu(h @ W1 + b1)
    gemm_bf16_k<<<gFF1, 256, 0, stream>>>(lnbuf, w1_bf + (size_t)l * D_ * FF_,
                                          nullptr, mid, b1 + l * FF_, nullptr,
                                          ROWS, FF_, 512, 1);
    // y2 = x2 + mid @ W2 + b2   (in place on x2)
    gemm_bf16_k<<<gProj, 256, 0, stream>>>(mid, w2_bf + (size_t)l * FF_ * D_,
                                           x2, nullptr, b2 + l * D_, x2,
                                           ROWS, 512, 2048, 0);
  }
  // final LN over concat(x1, x2)
  final_ln_k<<<ROWS, 256, 0, stream>>>(x1, x2, ln_o_s, ln_o_b, out);
}
